// GNNBase_28578712388145
// MI455X (gfx1250) — matrix operand order repaired and checked by reading, not compiled, not run on hardware
//
#include <hip/hip_runtime.h>
#include <hip/hip_bf16.h>

typedef __attribute__((ext_vector_type(16))) _Float16 v16h;
typedef __attribute__((ext_vector_type(8)))  _Float16 v8h;
typedef __attribute__((ext_vector_type(8)))  float    v8f;
typedef __attribute__((ext_vector_type(4)))  float    v4f;

__device__ inline v16h cat8(v8h lo, v8h hi) {
  return __builtin_shufflevector(lo, hi, 0, 1, 2, 3, 4, 5, 6, 7, 8, 9, 10, 11,
                                 12, 13, 14, 15);
}

// ---------------------------------------------------------------------------
// WMMA fragment helpers (gfx1250 wave32, v_wmma_f32_16x16x32_f16)
// A (16x32 f16): lane&15 = row M; halves 0..7 -> K = 2i   + (lane>=16 ? 8:0)
//                            halves 8..15 -> K = 2i+8 + (lane>=16 ? 8:0)
// B (32x16 f16): lane&15 = col N; same K mapping as A
// C/D (16x16 f32): vgpr v -> M = v + (lane>=16 ? 8:0); N = lane&15
// All fragment chunks are 16B aligned -> 128-bit loads.
// ---------------------------------------------------------------------------

// A fragment from an LDS tile [64][128]
__device__ inline v16h frag_a_sh(const _Float16 (&buf)[64 * 128], int row,
                                 int kbase, int lane) {
  const int base = row * 128 + kbase + ((lane & 16) ? 8 : 0);
  v8h lo = *(const v8h*)&buf[base];
  v8h hi = *(const v8h*)&buf[base + 16];
  return cat8(lo, hi);
}

// B fragment from LDS-staged transposed weights w[col][k] (ldk = KP)
__device__ inline v16h frag_b_sh(const _Float16 (&w)[128 * 128], int ldk,
                                 int col, int kbase, int lane) {
  const int base = col * ldk + kbase + ((lane & 16) ? 8 : 0);
  v8h lo = *(const v8h*)&w[base];
  v8h hi = *(const v8h*)&w[base + 16];
  return cat8(lo, hi);
}

// A fragment from global f16 (row-major, stride ld; ld multiple of 8)
__device__ inline v16h frag_a_f16(const _Float16* __restrict__ buf, int ld,
                                  int row, int kbase, int lane) {
  const _Float16* p =
      buf + (size_t)row * ld + kbase + ((lane & 16) ? 8 : 0);
  v8h lo = *(const v8h*)p;
  v8h hi = *(const v8h*)(p + 16);
  return cat8(lo, hi);
}

// A fragment from global f32 (converted to f16 on the fly)
__device__ inline v16h frag_a_f32(const float* __restrict__ buf, int ld,
                                  int row, int kbase, int lane) {
  const float* p = buf + (size_t)row * ld + kbase + ((lane & 16) ? 8 : 0);
  v4f q0 = *(const v4f*)p;
  v4f q1 = *(const v4f*)(p + 4);
  v4f q2 = *(const v4f*)(p + 16);
  v4f q3 = *(const v4f*)(p + 20);
  v16h a;
#pragma unroll
  for (int i = 0; i < 4; ++i) {
    a[i] = (_Float16)q0[i];
    a[i + 4] = (_Float16)q1[i];
    a[i + 8] = (_Float16)q2[i];
    a[i + 12] = (_Float16)q3[i];
  }
  return a;
}

// B fragment from global transposed weights wt[col][k]
__device__ inline v16h frag_b_wt(const _Float16* __restrict__ wt, int ldk,
                                 int col, int kbase, int lane) {
  const _Float16* p = wt + (size_t)col * ldk + kbase + ((lane & 16) ? 8 : 0);
  v8h lo = *(const v8h*)p;
  v8h hi = *(const v8h*)(p + 16);
  return cat8(lo, hi);
}

// branch-free tanh: one v_exp_f32, no EXEC divergence
__device__ inline float fast_tanh(float x) {
  x = fminf(fmaxf(x, -15.f), 15.f);
  float e = __expf(2.f * x);
  return (e - 1.f) / (e + 1.f);
}

// ---------------------------------------------------------------------------
// Prep kernels
// ---------------------------------------------------------------------------
__global__ __launch_bounds__(256) void wtrans_kernel(
    const float* __restrict__ W, _Float16* __restrict__ Wt, int K, int Ncols,
    int KP) {
  int idx = blockIdx.x * blockDim.x + threadIdx.x;
  if (idx >= Ncols * KP) return;
  int n = idx / KP, k = idx % KP;
  Wt[idx] = (k < K) ? (_Float16)W[(size_t)k * Ncols + n] : (_Float16)0.f;
}

// h0[n][0..62]=x[n][1..63]; [63..64]=embed[type]; [65..95]=0   (f16, stride 96)
__global__ __launch_bounds__(256) void build_h0_kernel(
    const float* __restrict__ x, const float* __restrict__ embed,
    _Float16* __restrict__ h0, int Nn) {
  int idx = blockIdx.x * blockDim.x + threadIdx.x;
  if (idx >= Nn * 96) return;
  int n = idx / 96, d = idx % 96;
  float v = 0.f;
  if (d < 63) {
    v = x[(size_t)n * 64 + d + 1];
  } else if (d < 65) {
    int t = (int)x[(size_t)n * 64];
    v = embed[t * 2 + (d - 63)];
  }
  h0[idx] = (_Float16)v;
}

__global__ __launch_bounds__(256) void init_nodes_kernel(
    float* __restrict__ lmax, float* __restrict__ denom, int Nn) {
  int i = blockIdx.x * blockDim.x + threadIdx.x;
  if (i < Nn) {
    lmax[i] = -INFINITY;
    denom[i] = 0.f;
  }
}

__global__ __launch_bounds__(256) void zero_f32_kernel(float* __restrict__ p,
                                                       long n) {
  long i = (long)blockIdx.x * blockDim.x + threadIdx.x;
  if (i < n) p[i] = 0.f;
}

// ---------------------------------------------------------------------------
// Register-blocked wave GEMM on LDS tiles: 16 rows x 128 cols, K = KP.
// One A-fragment per K-step feeds 8 accumulators. B fetch is software-
// pipelined one tile ahead so DS latency overlaps WMMA execution.
// ---------------------------------------------------------------------------
template <int KP>
__device__ inline void wave_gemm_tanh(const _Float16 (&inbuf)[64 * 128],
                                      _Float16 (&outbuf)[64 * 128],
                                      const _Float16 (&wlds)[128 * 128],
                                      const float* __restrict__ bias, int row0,
                                      int lane) {
  const int col0 = lane & 15;
  v8f acc[8];
  {
    v8f z = {};
#pragma unroll
    for (int nt = 0; nt < 8; ++nt) acc[nt] = z;
  }
#pragma unroll 1
  for (int kk = 0; kk < KP; kk += 32) {
    v16h a = frag_a_sh(inbuf, row0 + col0, kk, lane);
    v16h b = frag_b_sh(wlds, KP, col0, kk, lane);
#pragma unroll
    for (int nt = 0; nt < 8; ++nt) {
      v16h bn = b;
      if (nt < 7) bn = frag_b_sh(wlds, KP, (nt + 1) * 16 + col0, kk, lane);
      acc[nt] = __builtin_amdgcn_wmma_f32_16x16x32_f16(
          false, a, false, b, (short)0, acc[nt], false, false);
      b = bn;
    }
  }
  const int rbase = row0 + ((lane & 16) ? 8 : 0);
#pragma unroll
  for (int nt = 0; nt < 8; ++nt) {
    const int col = nt * 16 + col0;
    const float bv = bias[col];
#pragma unroll
    for (int v = 0; v < 8; ++v)
      outbuf[(rbase + v) * 128 + col] = (_Float16)fast_tanh(acc[nt][v] + bv);
  }
}

// cooperative 16B-chunk copy of a weight matrix into LDS
__device__ inline void stage_weights(const _Float16* __restrict__ g,
                                     _Float16 (&lds)[128 * 128], int halves,
                                     int tid, int nthreads) {
  const int4* src = (const int4*)g;
  const int n16 = halves >> 3;
  for (int i = tid; i < n16; i += nthreads) {
    int4 v = src[i];
    *(int4*)&lds[i * 8] = v;
  }
}

// ---------------------------------------------------------------------------
// Gate MLP over edges: m = [h[src][:NODE_PART], edge_attr] -> logits
// 64 edges per block, 4 waves, each wave owns a 16-edge row tile.
// ---------------------------------------------------------------------------
template <int DINP, int NODE_PART, int DREAL>
__global__ __launch_bounds__(128) void gate_kernel(
    const _Float16* __restrict__ h, int hstride,
    const float* __restrict__ edge_attr, const int* __restrict__ esrc,
    const _Float16* __restrict__ g0t, const float* __restrict__ b0,
    const _Float16* __restrict__ g1t, const float* __restrict__ b1,
    const _Float16* __restrict__ g2t, const float* __restrict__ b2,
    const float* __restrict__ w4, const float* __restrict__ b4,
    float* __restrict__ logits, int Etot) {
  __shared__ _Float16 bufA[64 * 128];
  __shared__ _Float16 bufB[64 * 128];
  __shared__ _Float16 wlds[128 * 128];
  const int tid = threadIdx.x;
  const int lane = tid & 31;
  const int wave = tid >> 5;
  const int e0 = blockIdx.x * 64;

  {  // stage messages into LDS (2 threads per edge) + first weight matrix
    const int e = tid >> 1;
    const int half = tid & 1;
    const int ge = min(e0 + e, Etot - 1);
    const int src = esrc[ge];
    const int dlo = half * (DINP / 2), dhi = dlo + DINP / 2;
    for (int d = dlo; d < dhi; ++d) {
      float v = 0.f;
      if (d < NODE_PART)
        v = (float)h[(size_t)src * hstride + d];
      else if (d < DREAL)
        v = edge_attr[(size_t)ge * 16 + (d - NODE_PART)];
      bufA[e * 128 + d] = (_Float16)v;
    }
    stage_weights(g0t, wlds, 128 * DINP, tid, 128);
  }
  __syncthreads();
  const int row0 = wave * 16;
  wave_gemm_tanh<DINP>(bufA, bufB, wlds, b0, row0, lane);
  __syncthreads();
  stage_weights(g1t, wlds, 128 * 128, tid, 128);
  __syncthreads();
  wave_gemm_tanh<128>(bufB, bufA, wlds, b1, row0, lane);
  __syncthreads();
  stage_weights(g2t, wlds, 128 * 128, tid, 128);
  __syncthreads();
  wave_gemm_tanh<128>(bufA, bufB, wlds, b2, row0, lane);
  __syncthreads();
  if (tid < 64) {
    const int ge = e0 + tid;
    if (ge < Etot) {
      float s = b4[0];
#pragma unroll 8
      for (int k = 0; k < 128; ++k) s += (float)bufB[tid * 128 + k] * w4[k];
      logits[ge] = s;
    }
  }
}

// ---------------------------------------------------------------------------
// Segment softmax pieces
// ---------------------------------------------------------------------------
__device__ inline void atomicMaxFloat(float* addr, float val) {
  if (val >= 0.f)
    atomicMax((int*)addr, __float_as_int(val));
  else
    atomicMin((unsigned int*)addr, (unsigned int)__float_as_int(val));
}

__global__ __launch_bounds__(256) void seg_max_kernel(
    const float* __restrict__ logits, const int* __restrict__ edst,
    float* __restrict__ lmax, int Etot) {
  int e = blockIdx.x * blockDim.x + threadIdx.x;
  if (e < Etot) atomicMaxFloat(&lmax[edst[e]], logits[e]);
}

__global__ __launch_bounds__(256) void exp_kernel(
    const float* __restrict__ logits, const int* __restrict__ edst,
    const float* __restrict__ lmax, float* __restrict__ ebuf,
    float* __restrict__ denom, int Etot) {
  int e = blockIdx.x * blockDim.x + threadIdx.x;
  if (e >= Etot) return;
  int d = edst[e];
  float lm = lmax[d];
  if (__builtin_isinf(lm)) lm = 0.f;
  float ev = __expf(logits[e] - lm);
  ebuf[e] = ev;
  atomicAdd(&denom[d], ev);
}

// scatter alpha * m into aggr (one thread per (edge, feature))
template <int DINP, int NODE_PART, int DREAL>
__global__ __launch_bounds__(256) void aggr_kernel(
    const _Float16* __restrict__ h, int hstride,
    const float* __restrict__ edge_attr, const int* __restrict__ esrc,
    const int* __restrict__ edst, const float* __restrict__ ebuf,
    const float* __restrict__ denom, float* __restrict__ aggr, int Etot) {
  long idx = (long)blockIdx.x * blockDim.x + threadIdx.x;
  int e = (int)(idx / DINP);
  int d = (int)(idx % DINP);
  if (e >= Etot || d >= DREAL) return;
  int dn = edst[e];
  float alpha = ebuf[e] / (denom[dn] + 1e-16f);
  float m;
  if (d < NODE_PART)
    m = (float)h[(size_t)esrc[e] * hstride + d];
  else
    m = edge_attr[(size_t)e * 16 + (d - NODE_PART)];
  atomicAdd(&aggr[(size_t)dn * DINP + d], alpha * m);
}

// ---------------------------------------------------------------------------
// Node update: out = aggr @ Wl + bl + h @ Wr  (WMMA, 16-node tiles per wave)
// ---------------------------------------------------------------------------
template <int KP, bool RELU, bool OUTF32>
__global__ __launch_bounds__(128) void node_linear_kernel(
    const float* __restrict__ aggr, const _Float16* __restrict__ h, int hstride,
    const _Float16* __restrict__ llt, const float* __restrict__ bl,
    const _Float16* __restrict__ lrt, _Float16* __restrict__ hout,
    float* __restrict__ fout, int Nn) {
  const int lane = threadIdx.x & 31;
  const int wave = blockIdx.x * (blockDim.x >> 5) + (threadIdx.x >> 5);
  const int row0 = wave * 16;
  if (row0 >= Nn) return;  // wave-uniform; EXEC stays full for WMMA
  const int r = min(row0 + (lane & 15), Nn - 1);
  const int off = (lane & 16) ? 8 : 0;
  const int col0 = lane & 15;
  v8f acc[8];
  {
    v8f z = {};
#pragma unroll
    for (int nt = 0; nt < 8; ++nt) acc[nt] = z;
  }
#pragma unroll 1
  for (int kk = 0; kk < KP; kk += 32) {
    v16h a1 = frag_a_f32(aggr, KP, r, kk, lane);
    v16h a2 = frag_a_f16(h, hstride, r, kk, lane);
#pragma unroll
    for (int nt = 0; nt < 8; ++nt) {
      const int col = nt * 16 + col0;
      v16h b1 = frag_b_wt(llt, KP, col, kk, lane);
      acc[nt] = __builtin_amdgcn_wmma_f32_16x16x32_f16(
          false, a1, false, b1, (short)0, acc[nt], false, false);
      v16h b2 = frag_b_wt(lrt, KP, col, kk, lane);
      acc[nt] = __builtin_amdgcn_wmma_f32_16x16x32_f16(
          false, a2, false, b2, (short)0, acc[nt], false, false);
    }
  }
#pragma unroll
  for (int nt = 0; nt < 8; ++nt) {
    const int col = nt * 16 + col0;
    const float bv = bl[col];
#pragma unroll
    for (int v = 0; v < 8; ++v) {
      int rr = row0 + off + v;
      if (rr < Nn) {
        float val = acc[nt][v] + bv;
        if (RELU) val = fmaxf(val, 0.f);
        if (OUTF32)
          fout[(size_t)rr * 128 + col] = val;
        else
          hout[(size_t)rr * 128 + col] = (_Float16)val;
      }
    }
  }
}

// ---------------------------------------------------------------------------
// Host driver
// ---------------------------------------------------------------------------
extern "C" void kernel_launch(void* const* d_in, const int* in_sizes, int n_in,
                              void* d_out, int out_size, void* d_ws,
                              size_t ws_size, hipStream_t stream) {
  const int Nn = in_sizes[0] / 64;
  const int Etot = in_sizes[2] / 2;

  const float* x = (const float*)d_in[0];
  const float* edge_attr = (const float*)d_in[1];
  const int* eidx = (const int*)d_in[2];
  const int* esrc = eidx;
  const int* edst = eidx + Etot;
  const float* embed = (const float*)d_in[3];
  // layer0: gate w/b x4, lin_l w/b, lin_r w ; layer1 same
  const float* L0w[4] = {(const float*)d_in[4], (const float*)d_in[6],
                         (const float*)d_in[8], (const float*)d_in[10]};
  const float* L0b[4] = {(const float*)d_in[5], (const float*)d_in[7],
                         (const float*)d_in[9], (const float*)d_in[11]};
  const float* L0llw = (const float*)d_in[12];
  const float* L0llb = (const float*)d_in[13];
  const float* L0lrw = (const float*)d_in[14];
  const float* L1w[4] = {(const float*)d_in[15], (const float*)d_in[17],
                         (const float*)d_in[19], (const float*)d_in[21]};
  const float* L1b[4] = {(const float*)d_in[16], (const float*)d_in[18],
                         (const float*)d_in[20], (const float*)d_in[22]};
  const float* L1llw = (const float*)d_in[23];
  const float* L1llb = (const float*)d_in[24];
  const float* L1lrw = (const float*)d_in[25];

  // --- workspace bump allocator ---
  char* wsb = (char*)d_ws;
  size_t off = 0;
  auto alloc = [&](size_t bytes) -> void* {
    void* p = wsb + off;
    off = (off + bytes + 255) & ~(size_t)255;
    return p;
  };
  _Float16* h0 = (_Float16*)alloc((size_t)Nn * 96 * 2);
  _Float16* h1 = (_Float16*)alloc((size_t)Nn * 128 * 2);
  float* logits = (float*)alloc((size_t)Etot * 4);
  float* ebuf = (float*)alloc((size_t)Etot * 4);
  float* lmax = (float*)alloc((size_t)Nn * 4);
  float* denom = (float*)alloc((size_t)Nn * 4);
  float* aggr = (float*)alloc((size_t)Nn * 128 * 4);
  _Float16* g0t0 = (_Float16*)alloc(128 * 96 * 2);
  _Float16* g1t0 = (_Float16*)alloc(128 * 128 * 2);
  _Float16* g2t0 = (_Float16*)alloc(128 * 128 * 2);
  _Float16* llt0 = (_Float16*)alloc(128 * 96 * 2);
  _Float16* lrt0 = (_Float16*)alloc(128 * 96 * 2);
  _Float16* g0t1 = (_Float16*)alloc(128 * 128 * 2);
  _Float16* g1t1 = (_Float16*)alloc(128 * 128 * 2);
  _Float16* g2t1 = (_Float16*)alloc(128 * 128 * 2);
  _Float16* llt1 = (_Float16*)alloc(128 * 128 * 2);
  _Float16* lrt1 = (_Float16*)alloc(128 * 128 * 2);

  auto grid1 = [](long n, int b) { return (unsigned)((n + b - 1) / b); };

  // --- weight prep (transpose + f16, K padded) ---
  wtrans_kernel<<<grid1(128 * 96, 256), 256, 0, stream>>>(L0w[0], g0t0, 81, 128, 96);
  wtrans_kernel<<<grid1(128 * 128, 256), 256, 0, stream>>>(L0w[1], g1t0, 128, 128, 128);
  wtrans_kernel<<<grid1(128 * 128, 256), 256, 0, stream>>>(L0w[2], g2t0, 128, 128, 128);
  wtrans_kernel<<<grid1(128 * 96, 256), 256, 0, stream>>>(L0llw, llt0, 81, 128, 96);
  wtrans_kernel<<<grid1(128 * 96, 256), 256, 0, stream>>>(L0lrw, lrt0, 81, 128, 96);
  wtrans_kernel<<<grid1(128 * 128, 256), 256, 0, stream>>>(L1w[0], g0t1, 128, 128, 128);
  wtrans_kernel<<<grid1(128 * 128, 256), 256, 0, stream>>>(L1w[1], g1t1, 128, 128, 128);
  wtrans_kernel<<<grid1(128 * 128, 256), 256, 0, stream>>>(L1w[2], g2t1, 128, 128, 128);
  wtrans_kernel<<<grid1(128 * 128, 256), 256, 0, stream>>>(L1llw, llt1, 128, 128, 128);
  wtrans_kernel<<<grid1(128 * 128, 256), 256, 0, stream>>>(L1lrw, lrt1, 128, 128, 128);

  build_h0_kernel<<<grid1((long)Nn * 96, 256), 256, 0, stream>>>(x, embed, h0, Nn);

  const unsigned eblocks = grid1(Etot, 64);
  const unsigned nwaveblk = grid1((Nn + 15) / 16, 4);

  // ================= layer 0 (DINP=96, NODE_PART=65, DREAL=81) ==============
  gate_kernel<96, 65, 81><<<eblocks, 128, 0, stream>>>(
      h0, 96, edge_attr, esrc, g0t0, L0b[0], g1t0, L0b[1], g2t0, L0b[2],
      L0w[3], L0b[3], logits, Etot);
  init_nodes_kernel<<<grid1(Nn, 256), 256, 0, stream>>>(lmax, denom, Nn);
  zero_f32_kernel<<<grid1((long)Nn * 96, 256), 256, 0, stream>>>(aggr, (long)Nn * 96);
  seg_max_kernel<<<grid1(Etot, 256), 256, 0, stream>>>(logits, edst, lmax, Etot);
  exp_kernel<<<grid1(Etot, 256), 256, 0, stream>>>(logits, edst, lmax, ebuf, denom, Etot);
  aggr_kernel<96, 65, 81><<<grid1((long)Etot * 96, 256), 256, 0, stream>>>(
      h0, 96, edge_attr, esrc, edst, ebuf, denom, aggr, Etot);
  node_linear_kernel<96, true, false><<<nwaveblk, 128, 0, stream>>>(
      aggr, h0, 96, llt0, L0llb, lrt0, h1, nullptr, Nn);

  // ================= layer 1 (DINP=128, NODE_PART=112, DREAL=128) ===========
  gate_kernel<128, 112, 128><<<eblocks, 128, 0, stream>>>(
      h1, 128, edge_attr, esrc, g0t1, L1b[0], g1t1, L1b[1], g2t1, L1b[2],
      L1w[3], L1b[3], logits, Etot);
  init_nodes_kernel<<<grid1(Nn, 256), 256, 0, stream>>>(lmax, denom, Nn);
  zero_f32_kernel<<<grid1((long)Nn * 128, 256), 256, 0, stream>>>(aggr, (long)Nn * 128);
  seg_max_kernel<<<grid1(Etot, 256), 256, 0, stream>>>(logits, edst, lmax, Etot);
  exp_kernel<<<grid1(Etot, 256), 256, 0, stream>>>(logits, edst, lmax, ebuf, denom, Etot);
  aggr_kernel<128, 112, 128><<<grid1((long)Etot * 128, 256), 256, 0, stream>>>(
      h1, 128, edge_attr, esrc, edst, ebuf, denom, aggr, Etot);
  node_linear_kernel<128, false, true><<<nwaveblk, 128, 0, stream>>>(
      aggr, h1, 128, llt1, L1llb, lrt1, nullptr, (float*)d_out, Nn);
}
